// TDSM_77996606095433
// MI455X (gfx1250) — matrix operand order repaired and checked by reading, not compiled, not run on hardware
//
#include <hip/hip_runtime.h>
#include <hip/hip_bf16.h>

#define B_   2
#define N_   4096
#define DIM_ 1152
#define E_   6
#define De_  192
#define H_   4
#define S_   4
#define HD_  48
#define L_   1024
#define EPS_ 1e-6f
#define MROWS 8192   /* B_*S_*L_ rows per expert */

typedef __bf16 bf16;
typedef __attribute__((ext_vector_type(16))) __bf16 v16bf;
typedef __attribute__((ext_vector_type(8)))  __bf16 v8bf;
typedef __attribute__((ext_vector_type(8)))  float  v8f;
typedef __attribute__((ext_vector_type(4)))  float  v4f;
typedef __attribute__((ext_vector_type(4)))  unsigned int v4u;
typedef __attribute__((ext_vector_type(4)))  int v4i;
typedef __attribute__((ext_vector_type(8)))  int v8i;

// Issue a 2D TDM load: tile_d0 x tile_d1 elements of 2 bytes, row stride
// `stride_elems`, from global `src` into LDS at byte offset `lds_addr`.
__device__ __forceinline__ void tdm_load_2d_bf16(unsigned lds_addr, const void* src,
                                                 unsigned tile_d0, unsigned tile_d1,
                                                 unsigned stride_elems) {
  unsigned long long ga = (unsigned long long)(size_t)src;
  v4u g0;
  g0[0] = 1u;                                   // count=1, user descriptor
  g0[1] = lds_addr;                             // lds_addr
  g0[2] = (unsigned)ga;                         // global_addr[31:0]
  g0[3] = (unsigned)((ga >> 32) & 0x01FFFFFFull) | 0x80000000u; // [56:32] | type=2
  v8i g1;
  g1[0] = (int)(1u << 16);                      // data_size=1 (2 bytes)
  g1[1] = (int)((stride_elems & 0xFFFFu) << 16);        // tensor_dim0[15:0]
  g1[2] = (int)(((unsigned)MROWS & 0xFFFFu) << 16) |
          (int)(stride_elems >> 16);            // tensor_dim0[31:16] | tensor_dim1[15:0]
  g1[3] = (int)((tile_d0 & 0xFFFFu) << 16) | (int)((unsigned)MROWS >> 16); // tile_dim0
  g1[4] = (int)(tile_d1 & 0xFFFFu);             // tile_dim1 (tile_dim2=0)
  g1[5] = (int)stride_elems;                    // tensor_dim0_stride[31:0]
  g1[6] = 0;                                    // stride0[47:32] | stride1[15:0]
  g1[7] = 0;
  v4i g2 = {0, 0, 0, 0};
  v4i g3 = {0, 0, 0, 0};
#if defined(__clang_major__) && (__clang_major__ >= 23)
  v8i g4 = {0, 0, 0, 0, 0, 0, 0, 0};
  __builtin_amdgcn_tensor_load_to_lds(g0, g1, g2, g3, g4, 0);
#else
  __builtin_amdgcn_tensor_load_to_lds(g0, g1, g2, g3, 0);
#endif
}

// ---------------------------------------------------------------- stats
__global__ void k_stats(const float* __restrict__ c, float* __restrict__ stats) {
  int wid = threadIdx.x >> 5, lane = threadIdx.x & 31;
  int row = blockIdx.x * 8 + wid;            // 8 waves/block, wave per row
  const float* p = c + (size_t)row * DIM_;
  float s = 0.f, s2 = 0.f;
  for (int i = lane; i < DIM_; i += 32) { float v = p[i]; s += v; s2 += v * v; }
  #pragma unroll
  for (int off = 16; off > 0; off >>= 1) {
    s  += __shfl_xor(s,  off);
    s2 += __shfl_xor(s2, off);
  }
  if (lane == 0) {
    float mu  = s  * (1.f / DIM_);
    float var = s2 * (1.f / DIM_) - mu * mu;
    stats[row * 2 + 0] = mu;
    stats[row * 2 + 1] = rsqrtf(var + EPS_);
  }
}

// ------------------------------------------------- modulate + gather -> XS bf16
__global__ void k_gather(const float* __restrict__ c, const float* __restrict__ t,
                         const float* __restrict__ sst, const int* __restrict__ chperm,
                         const int* __restrict__ tokperm, const float* __restrict__ stats,
                         bf16* __restrict__ XS) {
  size_t idx = (size_t)blockIdx.x * blockDim.x + threadIdx.x;
  int d = (int)(idx % De_);
  size_t r = idx / De_;
  int l = (int)(r % L_); r /= L_;
  int s = (int)(r % S_); r /= S_;
  int b = (int)(r % B_);
  int e = (int)(r / B_);
  int j = (s * L_ + l) * De_ + d;
  int p = tokperm[(size_t)e * (N_ * De_) + j];
  int n = p / De_, dd = p - n * De_;
  int ch = chperm[e * De_ + dd];
  float mu = stats[(b * N_ + n) * 2 + 0];
  float rs = stats[(b * N_ + n) * 2 + 1];
  float shift = sst[ch]        + t[(size_t)b * 6 * DIM_ + ch];
  float scale = sst[DIM_ + ch] + t[(size_t)b * 6 * DIM_ + DIM_ + ch];
  float cv = c[((size_t)b * N_ + n) * DIM_ + ch];
  float v = (cv - mu) * rs * (1.f + scale) + shift;
  XS[idx] = (bf16)v;
}

// ------------------------------- NT GEMM: C[m,f] = sum_d A[m,d]*W[f,d] + bias[f]
// block: 128 threads (4 waves), 64x64 tile, K=192 in steps of 32, bf16 WMMA.
// A tile (bf16, strided rows) fetched by the Tensor Data Mover into LDS.
__global__ void k_gemm_nt(const bf16* __restrict__ A, const float* __restrict__ W,
                          const float* __restrict__ bias, bf16* __restrict__ Cout,
                          int Ncols) {
  __shared__ bf16 As[64][32];
  __shared__ bf16 Bs[64][32];
  int e  = blockIdx.z;
  int m0 = blockIdx.y * 64;
  int n0 = blockIdx.x * 64;
  const bf16*  Ae = A    + (size_t)e * MROWS * De_;
  const float* We = W    + (size_t)e * Ncols * De_;
  const float* be = bias + (size_t)e * Ncols;
  int tid = threadIdx.x, wid = tid >> 5, lane = tid & 31;
  int wm = (wid >> 1) * 32, wn = (wid & 1) * 32;
  int mrow = lane & 15, half = lane >> 4;
  unsigned as_lds = (unsigned)(size_t)&As[0][0];
  v8f acc[2][2];
  #pragma unroll
  for (int i = 0; i < 2; i++)
    #pragma unroll
    for (int j = 0; j < 2; j++)
      #pragma unroll
      for (int q = 0; q < 8; q++) acc[i][j][q] = 0.f;

  int lr = tid >> 1;           // 0..63
  int lc = (tid & 1) * 16;     // 0 or 16
  for (int kk = 0; kk < De_; kk += 32) {
    // stage A tile: 64 rows x 32 cols bf16 via TDM (one op from wave 0)
    if (tid < 32) {
      tdm_load_2d_bf16(as_lds, Ae + (size_t)m0 * De_ + kk, 32u, 64u, (unsigned)De_);
    }
    // stage B tile (fp32 -> bf16) with all threads
    {
      const float* src = We + (size_t)(n0 + lr) * De_ + kk + lc;
      #pragma unroll
      for (int i = 0; i < 16; i += 4) {
        v4f v = *(const v4f*)(src + i);
        Bs[lr][lc + i + 0] = (bf16)v.x; Bs[lr][lc + i + 1] = (bf16)v.y;
        Bs[lr][lc + i + 2] = (bf16)v.z; Bs[lr][lc + i + 3] = (bf16)v.w;
      }
    }
    if (tid < 32) __builtin_amdgcn_s_wait_tensorcnt((short)0);
    __syncthreads();
    int kb = half * 8, kb16 = half * 16;
    v16bf af[2], bfv[2];
    #pragma unroll
    for (int i = 0; i < 2; i++) {
      const bf16* pa = &As[wm + i * 16 + mrow][0];
      #pragma unroll
      for (int q = 0; q < 8; q++) { af[i][q] = pa[kb16 + q]; af[i][q + 8] = pa[kb16 + 8 + q]; }
    }
    #pragma unroll
    for (int j = 0; j < 2; j++)
      bfv[j] = *(const v16bf*)&Bs[wn + j * 16 + mrow][kb16];
    #pragma unroll
    for (int i = 0; i < 2; i++)
      #pragma unroll
      for (int j = 0; j < 2; j++)
        acc[i][j] = __builtin_amdgcn_wmma_f32_16x16x32_bf16(
            false, af[i], false, bfv[j], (short)0, acc[i][j], false, false);
    __syncthreads();
  }
  // epilogue: C layout row = q + 8*half, col = lane&15
  #pragma unroll
  for (int i = 0; i < 2; i++)
    #pragma unroll
    for (int j = 0; j < 2; j++) {
      int col = n0 + wn + j * 16 + mrow;
      float bv = be[col];
      #pragma unroll
      for (int q = 0; q < 8; q++) {
        int row = m0 + wm + i * 16 + q + 8 * half;
        Cout[((size_t)e * MROWS + row) * Ncols + col] = (bf16)(acc[i][j][q] + bv);
      }
    }
}

// ------------------------------------------- flash attention per (e,b,s,h)
// grid: (L/128, H, E*B*S); block 256 = 8 waves; wave owns 16 query rows
__global__ void k_attn(const bf16* __restrict__ QKV, bf16* __restrict__ O) {
  __shared__ bf16 Ks[64][64];       // [key][hd padded to 64]
  __shared__ bf16 Vt[48][64];       // [hd][key]
  __shared__ bf16 Ps[8][16][16];    // per-wave P staging
  int rb = blockIdx.x, h = blockIdx.y, ebs = blockIdx.z;
  int tid = threadIdx.x, wid = tid >> 5, lane = tid & 31;
  size_t rowbase = (size_t)ebs * L_;
  int base_l = rb * 128 + wid * 16;
  int mrow = lane & 15, half = lane >> 4;
  int kb = half * 8;

  // zero the hd 48..63 pad region of Ks once
  for (int i = tid; i < 64 * 16; i += 256) Ks[i >> 4][48 + (i & 15)] = (bf16)0.f;

  // Q fragments (A-matrix), HD 48 padded to 64 -> two k-steps
  const bf16* qrow = QKV + (rowbase + base_l + mrow) * 576 + h * HD_;
  v16bf a0, a1;
  #pragma unroll
  for (int q = 0; q < 8; q++) {
    a0[q]     = qrow[kb + q];
    a0[q + 8] = qrow[kb + 16 + q];
    a1[q]     = qrow[32 + kb + q];   // hd 32..47
    a1[q + 8] = (bf16)0.f;           // hd 48..63 pad
  }

  float mrun[8], lrun[8];
  v8f o[3];
  #pragma unroll
  for (int q = 0; q < 8; q++) { mrun[q] = -1e30f; lrun[q] = 0.f; o[0][q] = 0.f; o[1][q] = 0.f; o[2][q] = 0.f; }
  const float sm = 0.14433756729740643f;   // 1/sqrt(48)

  // staging task mapping: 192 active threads, chunk cc = tid>>6, key = tid&63
  int cc = tid >> 6, keyt = tid & 63;
  for (int kc = 0; kc < L_; kc += 64) {
    if (tid < 192) {
      const bf16* kr = QKV + (rowbase + kc + keyt) * 576 + De_ + h * HD_ + cc * 16;
      *(v16bf*)&Ks[keyt][cc * 16] = *(const v16bf*)kr;   // 32B copy
      const bf16* vr = QKV + (rowbase + kc + keyt) * 576 + 2 * De_ + h * HD_ + cc * 16;
      v16bf vv = *(const v16bf*)vr;
      #pragma unroll
      for (int i = 0; i < 16; i++) Vt[cc * 16 + i][keyt] = vv[i];  // transpose store
    }
    __syncthreads();
    #pragma unroll
    for (int kt = 0; kt < 4; kt++) {
      // S = Q * K^T  (B-fragment: col = key, K = hd)
      v8bf b0l = *(const v8bf*)&Ks[kt * 16 + mrow][kb];
      v8bf b0h = *(const v8bf*)&Ks[kt * 16 + mrow][kb + 16];
      v8bf b1l = *(const v8bf*)&Ks[kt * 16 + mrow][32 + kb];
      v8bf b1h = *(const v8bf*)&Ks[kt * 16 + mrow][48 + kb];
      v16bf b0 = __builtin_shufflevector(b0l, b0h, 0, 1, 2, 3, 4, 5, 6, 7, 8, 9, 10, 11, 12, 13, 14, 15);
      v16bf b1 = __builtin_shufflevector(b1l, b1h, 0, 1, 2, 3, 4, 5, 6, 7, 8, 9, 10, 11, 12, 13, 14, 15);
      v8f sacc;
      #pragma unroll
      for (int q = 0; q < 8; q++) sacc[q] = 0.f;
      sacc = __builtin_amdgcn_wmma_f32_16x16x32_bf16(false, a0, false, b0, (short)0, sacc, false, false);
      sacc = __builtin_amdgcn_wmma_f32_16x16x32_bf16(false, a1, false, b1, (short)0, sacc, false, false);
      float pv[8];
      #pragma unroll
      for (int q = 0; q < 8; q++) {
        float x = sacc[q] * sm;
        float rm = x;
        rm = fmaxf(rm, __shfl_xor(rm, 1)); rm = fmaxf(rm, __shfl_xor(rm, 2));
        rm = fmaxf(rm, __shfl_xor(rm, 4)); rm = fmaxf(rm, __shfl_xor(rm, 8));
        float mn = fmaxf(mrun[q], rm);
        float alpha = __expf(mrun[q] - mn);
        float p = __expf(x - mn);
        float rs = p;
        rs += __shfl_xor(rs, 1); rs += __shfl_xor(rs, 2);
        rs += __shfl_xor(rs, 4); rs += __shfl_xor(rs, 8);
        lrun[q] = lrun[q] * alpha + rs;
        mrun[q] = mn;
        o[0][q] *= alpha; o[1][q] *= alpha; o[2][q] *= alpha;
        pv[q] = p;
      }
      // C-layout -> A-layout via per-wave LDS (same-wave in-order DS)
      #pragma unroll
      for (int q = 0; q < 8; q++) Ps[wid][q + 8 * half][mrow] = (bf16)pv[q];
      v16bf pf;
      const bf16* pp = &Ps[wid][mrow][0];
      #pragma unroll
      for (int q = 0; q < 8; q++) { pf[q] = half ? (bf16)0.f : pp[q]; pf[q + 8] = half ? (bf16)0.f : pp[8 + q]; }
      // O += P * V  (B-fragment: col = hd, K = key; A zero beyond K=15)
      #pragma unroll
      for (int nt = 0; nt < 3; nt++) {
        v16bf vb = *(const v16bf*)&Vt[nt * 16 + mrow][kt * 16];
        o[nt] = __builtin_amdgcn_wmma_f32_16x16x32_bf16(false, pf, false, vb, (short)0, o[nt], false, false);
      }
    }
    __syncthreads();
  }
  #pragma unroll
  for (int nt = 0; nt < 3; nt++)
    #pragma unroll
    for (int q = 0; q < 8; q++) {
      int lr = base_l + q + 8 * half;
      float vo = o[nt][q] / lrun[q];
      O[(rowbase + lr) * De_ + h * HD_ + nt * 16 + mrow] = (bf16)vo;
    }
}

// --------------------------- inverse perms + gated residual -> fp32 out
__global__ void k_scatter(const float* __restrict__ c, const float* __restrict__ t,
                          const float* __restrict__ sst, const int* __restrict__ chperm,
                          const int* __restrict__ tokperm, const bf16* __restrict__ P,
                          float* __restrict__ out) {
  size_t idx = (size_t)blockIdx.x * blockDim.x + threadIdx.x;
  int d = (int)(idx % De_);
  size_t r = idx / De_;
  int l = (int)(r % L_); r /= L_;
  int s = (int)(r % S_); r /= S_;
  int b = (int)(r % B_);
  int e = (int)(r / B_);
  int j = (s * L_ + l) * De_ + d;
  int p = tokperm[(size_t)e * (N_ * De_) + j];
  int n = p / De_, dd = p - n * De_;
  int ch = chperm[e * De_ + dd];
  float gate = sst[2 * DIM_ + ch] + t[(size_t)b * 6 * DIM_ + 2 * DIM_ + ch];
  float pv = (float)P[idx];
  size_t oi = ((size_t)b * N_ + n) * DIM_ + ch;
  out[oi] = c[oi] + gate * pv;
}

extern "C" void kernel_launch(void* const* d_in, const int* in_sizes, int n_in,
                              void* d_out, int out_size, void* d_ws, size_t ws_size,
                              hipStream_t stream) {
  const float* c      = (const float*)d_in[0];
  /* d_in[1] = y (unused by the reference computation) */
  const float* t      = (const float*)d_in[2];
  const float* sst    = (const float*)d_in[3];
  const float* qkv_w  = (const float*)d_in[4];
  const float* qkv_b  = (const float*)d_in[5];
  const float* proj_w = (const float*)d_in[6];
  const float* proj_b = (const float*)d_in[7];
  const int*   chperm = (const int*)d_in[8];
  const int*   tokperm= (const int*)d_in[9];
  float* out = (float*)d_out;

  char* ws = (char*)d_ws;
  size_t off = 0;
  float* stats = (float*)(ws + off); off += 65536;
  bf16*  XS    = (bf16*)(ws + off);  off += (size_t)E_ * MROWS * De_ * 2;
  bf16*  QKV   = (bf16*)(ws + off);  off += (size_t)E_ * MROWS * 3 * De_ * 2;
  bf16*  OATT  = (bf16*)(ws + off);  off += (size_t)E_ * MROWS * De_ * 2;
  bf16*  PROJ  = (bf16*)(ws + off);

  k_stats  <<<dim3(B_ * N_ / 8),             dim3(256), 0, stream>>>(c, stats);
  k_gather <<<dim3(E_ * MROWS * De_ / 256),  dim3(256), 0, stream>>>(c, t, sst, chperm, tokperm, stats, XS);
  k_gemm_nt<<<dim3(576 / 64, MROWS / 64, E_),dim3(128), 0, stream>>>(XS, qkv_w, qkv_b, QKV, 576);
  k_attn   <<<dim3(L_ / 128, H_, E_ * B_ * S_), dim3(256), 0, stream>>>(QKV, OATT);
  k_gemm_nt<<<dim3(192 / 64, MROWS / 64, E_),dim3(128), 0, stream>>>(OATT, proj_w, proj_b, PROJ, 192);
  k_scatter<<<dim3(E_ * MROWS * De_ / 256),  dim3(256), 0, stream>>>(c, t, sst, chperm, tokperm, PROJ, out);
}
